// PointnetFP_52750788329675
// MI455X (gfx1250) — matrix operand order repaired and checked by reading, not compiled, not run on hardware
//
#include <hip/hip_runtime.h>
#include <hip/hip_bf16.h>

// ---------------------------------------------------------------------------
// Problem constants (from setup_inputs: B=4, N1=8192, N2=2048, C1=128, C2=256)
// ---------------------------------------------------------------------------
#define BB   4
#define N1   8192
#define N2   2048
#define C1   128
#define C2   256
#define F    256          // output features of both MLP layers
#define K1   (C2 + C1)    // 384
#define M    (BB * N1)    // 32768 rows through the MLP

// Workspace layout (byte offsets; all 16B-aligned)
#define OFF_IDX  0u                                   // int   [M][3]      393216 B
#define OFF_WGT  (OFF_IDX + (size_t)M * 3 * 4)        // float [M][3]      393216 B
#define OFF_X    (OFF_WGT + (size_t)M * 3 * 4)        // bf16  [M][384]  25165824 B
#define OFF_W1P  (OFF_X   + (size_t)M * K1 * 2)       // bf16  [12][256][32] 196608 B
#define OFF_W2P  (OFF_W1P + (size_t)K1 * F * 2)       // bf16  [8][256][32]  131072 B
#define OFF_H    (OFF_W2P + (size_t)F * F * 2)        // bf16  [M][256]  16777216 B
// total ~41 MB

typedef __attribute__((ext_vector_type(16))) __bf16        v16bf;
typedef __attribute__((ext_vector_type(8)))  float         v8f;
typedef __attribute__((ext_vector_type(4)))  unsigned int  u32x4;

union Frag { v16bf v; u32x4 q[2]; };

static __device__ __forceinline__ unsigned short f32_to_bf16(float f) {
    unsigned int u = __float_as_uint(f);
    // round-to-nearest-even
    u += 0x7FFFu + ((u >> 16) & 1u);
    return (unsigned short)(u >> 16);
}

// ---------------------------------------------------------------------------
// Kernel 1: brute-force 3-NN + inverse-distance weights.
// One thread per query point; xyz2 tiles staged through LDS.
// grid = B * (N1/256), block = 256
// ---------------------------------------------------------------------------
__global__ __launch_bounds__(256) void three_nn_kernel(
    const float* __restrict__ xyz1, const float* __restrict__ xyz2,
    int* __restrict__ idx, float* __restrict__ wgt)
{
    __shared__ float sx[256], sy[256], sz[256];

    const int b     = blockIdx.x >> 5;          // N1/256 == 32 chunks per batch
    const int chunk = blockIdx.x & 31;
    const int q     = (chunk << 8) + threadIdx.x;

    const float* p = xyz1 + ((size_t)b * N1 + q) * 3;
    const float x = p[0], y = p[1], z = p[2];

    float d0 = 3.4e38f, d1 = 3.4e38f, d2 = 3.4e38f;
    int   i0 = 0, i1 = 0, i2 = 0;

    for (int t = 0; t < N2; t += 256) {
        const float* rp = xyz2 + ((size_t)b * N2 + t + threadIdx.x) * 3;
        sx[threadIdx.x] = rp[0];
        sy[threadIdx.x] = rp[1];
        sz[threadIdx.x] = rp[2];
        __syncthreads();
        #pragma unroll 8
        for (int j = 0; j < 256; ++j) {
            const float dx = x - sx[j], dy = y - sy[j], dz = z - sz[j];
            const float d = dx * dx + dy * dy + dz * dz;
            const int jj = t + j;
            if (d < d2) {
                if (d < d1) {
                    if (d < d0) { d2 = d1; i2 = i1; d1 = d0; i1 = i0; d0 = d; i0 = jj; }
                    else        { d2 = d1; i2 = i1; d1 = d;  i1 = jj; }
                } else          { d2 = d;  i2 = jj; }
            }
        }
        __syncthreads();
    }

    d0 = fmaxf(d0, 1e-10f); d1 = fmaxf(d1, 1e-10f); d2 = fmaxf(d2, 1e-10f);
    float w0 = 1.0f / d0, w1 = 1.0f / d1, w2 = 1.0f / d2;
    const float inv = 1.0f / (w0 + w1 + w2);
    w0 *= inv; w1 *= inv; w2 *= inv;

    const size_t r = (size_t)b * N1 + q;
    idx[r * 3 + 0] = i0; idx[r * 3 + 1] = i1; idx[r * 3 + 2] = i2;
    wgt[r * 3 + 0] = w0; wgt[r * 3 + 1] = w1; wgt[r * 3 + 2] = w2;
}

// ---------------------------------------------------------------------------
// Kernel 2: repack fp32 weights [K][256] into WMMA-B-friendly bf16 layout
// Wp[kt][n][kk] with kt=k/32, kk=k%32 (so each column's 32 K-values are
// contiguous -> one 32-byte fragment load per lane).
// grid = K/32, block = 256
// ---------------------------------------------------------------------------
__global__ __launch_bounds__(256) void pack_w_kernel(
    const float* __restrict__ w, unsigned short* __restrict__ wp)
{
    const int kt = blockIdx.x;
    const int n  = threadIdx.x;
    #pragma unroll
    for (int kk = 0; kk < 32; ++kk)
        wp[((size_t)kt * 256 + n) * 32 + kk] =
            f32_to_bf16(w[(size_t)(kt * 32 + kk) * 256 + n]);
}

// ---------------------------------------------------------------------------
// Kernel 3: build X = concat(interpolated points2 features, points1) as bf16
// row-major [M][384]. One block per output row.
// grid = M, block = 384
// ---------------------------------------------------------------------------
__global__ __launch_bounds__(384) void build_x_kernel(
    const float* __restrict__ points1, const float* __restrict__ points2,
    const int* __restrict__ idx, const float* __restrict__ wgt,
    unsigned short* __restrict__ X)
{
    __shared__ int   si[3];
    __shared__ float sw[3];

    const int r = blockIdx.x;            // global row: b*N1 + n
    const int b = r >> 13;               // / 8192
    const int c = threadIdx.x;

    if (c < 3) { si[c] = idx[(size_t)r * 3 + c]; sw[c] = wgt[(size_t)r * 3 + c]; }
    __syncthreads();

    float v;
    if (c < C2) {
        const float* P2 = points2 + (size_t)b * N2 * C2;
        v = sw[0] * P2[(size_t)si[0] * C2 + c]
          + sw[1] * P2[(size_t)si[1] * C2 + c]
          + sw[2] * P2[(size_t)si[2] * C2 + c];
    } else {
        v = points1[(size_t)r * C1 + (c - C2)];
    }
    X[(size_t)r * K1 + c] = f32_to_bf16(v);
}

// ---------------------------------------------------------------------------
// Kernel 4/5: GEMM via v_wmma_f32_16x16x32_bf16 + ReLU.
//   A  : bf16 row-major [M][K]      (A-fragment = two contiguous b128 loads)
//   Bp : bf16 packed   [K/32][256][32] (B-fragment = one contiguous 32B load)
// Block = 256 threads (8 waves). Block tile = 16 rows x 256 cols; each wave
// owns two 16x16 N-tiles. K fully unrolled -> 2*(K/32) WMMAs per wave.
// grid = M/16, block = 256
// ---------------------------------------------------------------------------
template <int K, bool OUT_F32>
__global__ __launch_bounds__(256) void gemm_wmma_kernel(
    const unsigned short* __restrict__ A,
    const unsigned short* __restrict__ Bp,
    unsigned short* __restrict__ Ob,   // bf16 out (OUT_F32 == false)
    float* __restrict__ Of)            // f32 out  (OUT_F32 == true)
{
    const int m0   = blockIdx.x << 4;
    const int lane = threadIdx.x & 31;
    const int wave = threadIdx.x >> 5;
    const int h    = lane >> 4;       // half-wave select
    const int c    = lane & 15;       // row (A) / col (B,D) within tile
    const int n0   = wave << 5;       // this wave's first output column

    const unsigned short* arow = A + (size_t)(m0 + c) * K;

    v8f acc0 = {}, acc1 = {};

    #pragma unroll
    for (int kt = 0; kt < K / 32; ++kt) {
        Frag a, b0, b1;
        // A fragment: lane holds row c, K = 16h..16h+15
        a.q[0] = *(const u32x4*)(arow + kt * 32 + 16 * h);
        a.q[1] = *(const u32x4*)(arow + kt * 32 + 16 * h + 8);
        // B fragments: lane holds col n, K = 16h..16h+15 (contiguous packed)
        const unsigned short* bb = Bp + (size_t)kt * 256 * 32;
        const unsigned short* c0 = bb + ((n0 + c) << 5) + (h << 4);
        const unsigned short* c1 = bb + ((n0 + 16 + c) << 5) + (h << 4);
        b0.q[0] = *(const u32x4*)(c0);
        b0.q[1] = *(const u32x4*)(c0 + 8);
        b1.q[0] = *(const u32x4*)(c1);
        b1.q[1] = *(const u32x4*)(c1 + 8);

        acc0 = __builtin_amdgcn_wmma_f32_16x16x32_bf16(
            false, a.v, false, b0.v, (short)0, acc0, false, false);
        acc1 = __builtin_amdgcn_wmma_f32_16x16x32_bf16(
            false, a.v, false, b1.v, (short)0, acc1, false, false);
    }

    // D layout: acc element j -> row m0 + j + 8h, col n0 + c (and +16)
    #pragma unroll
    for (int j = 0; j < 8; ++j) {
        const int row = m0 + j + (h << 3);
        const float v0 = fmaxf(acc0[j], 0.0f);
        const float v1 = fmaxf(acc1[j], 0.0f);
        if (OUT_F32) {
            Of[(size_t)row * F + n0 + c]      = v0;
            Of[(size_t)row * F + n0 + 16 + c] = v1;
        } else {
            Ob[(size_t)row * F + n0 + c]      = f32_to_bf16(v0);
            Ob[(size_t)row * F + n0 + 16 + c] = f32_to_bf16(v1);
        }
    }
}

// ---------------------------------------------------------------------------
// Launch
// ---------------------------------------------------------------------------
extern "C" void kernel_launch(void* const* d_in, const int* in_sizes, int n_in,
                              void* d_out, int out_size, void* d_ws, size_t ws_size,
                              hipStream_t stream) {
    (void)in_sizes; (void)n_in; (void)out_size; (void)ws_size;

    const float* xyz1    = (const float*)d_in[0];   // [B,N1,3]
    const float* xyz2    = (const float*)d_in[1];   // [B,N2,3]
    const float* points1 = (const float*)d_in[2];   // [B,N1,C1]
    const float* points2 = (const float*)d_in[3];   // [B,N2,C2]
    const float* w1      = (const float*)d_in[4];   // [384,256]
    const float* w2      = (const float*)d_in[5];   // [256,256]
    float*       out     = (float*)d_out;           // [B,N1,256]

    char* ws = (char*)d_ws;
    int*            ridx = (int*)(ws + OFF_IDX);
    float*          rwgt = (float*)(ws + OFF_WGT);
    unsigned short* X    = (unsigned short*)(ws + OFF_X);
    unsigned short* W1p  = (unsigned short*)(ws + OFF_W1P);
    unsigned short* W2p  = (unsigned short*)(ws + OFF_W2P);
    unsigned short* H    = (unsigned short*)(ws + OFF_H);

    // 1) 3-NN + weights
    three_nn_kernel<<<dim3(BB * (N1 / 256)), dim3(256), 0, stream>>>(xyz1, xyz2, ridx, rwgt);

    // 2) weight repack (f32 -> bf16, WMMA-B layout)
    pack_w_kernel<<<dim3(K1 / 32), dim3(256), 0, stream>>>(w1, W1p);
    pack_w_kernel<<<dim3(F  / 32), dim3(256), 0, stream>>>(w2, W2p);

    // 3) interpolate + concat -> X bf16 [M][384]
    build_x_kernel<<<dim3(M), dim3(K1), 0, stream>>>(points1, points2, ridx, rwgt, X);

    // 4) H = relu(X @ W1)  (bf16 out)
    gemm_wmma_kernel<K1, false><<<dim3(M / 16), dim3(256), 0, stream>>>(X, W1p, H, nullptr);

    // 5) out = relu(H @ W2) (f32 out)
    gemm_wmma_kernel<F, true><<<dim3(M / 16), dim3(256), 0, stream>>>(H, W2p, nullptr, out);
}
